// GNNEncoder_10290741641368
// MI455X (gfx1250) — matrix operand order repaired and checked by reading, not compiled, not run on hardware
//
#include <hip/hip_runtime.h>
#include <stdint.h>
#include <stddef.h>

#define DEV __device__ __forceinline__

typedef __attribute__((ext_vector_type(16))) __bf16 v16bf;
typedef __attribute__((ext_vector_type(8)))  float  v8f;

namespace {
constexpr int Bc  = 64, Nc = 64, Dc = 128;
constexpr int Ec  = Nc * (Nc - 1);            // 4032 edges
constexpr int EHc = 256, EOc = 128, NHc = 256, NOc = 128;
constexpr long long BE  = (long long)Bc * Ec; // 258048 edge rows
constexpr int BNr = Bc * Nc;                  // 4096 node rows
constexpr int WAVES = 4;                      // wave32 waves per block
constexpr float BN_EPS = 1e-5f;
}

union FragBF { uint32_t u[8]; v16bf v; };

DEV uint16_t f2bf(float f) {
  uint32_t u = __builtin_bit_cast(uint32_t, f);
  u += 0x7FFFu + ((u >> 16) & 1u);            // round-to-nearest-even
  return (uint16_t)(u >> 16);
}
DEV float elu(float x) { return x > 0.f ? x : (__expf(x) - 1.f); }

// ---------------------------------------------------------------------------
// Two fused dense layers for one 16-row slab, bf16 WMMA, f32 accumulate.
//  xin : LDS bf16 tile [16][K1]   (this wave's private segment)
//  h1  : LDS bf16 tile [16][N1]   (this wave's private segment)
//  W1t : global bf16 [N1][K1]     (transposed weights, K contiguous)
//  W2t : global bf16 [N2][N1]
//  h2row: global f32, points at row r0, row stride = N2
//  bnSum/bnSq: LDS f32[N2] block accumulators
// ---------------------------------------------------------------------------
template <int K1, int N1, int N2>
DEV void mlp2_core(const uint16_t* __restrict__ xin,
                   uint16_t* __restrict__ h1,
                   const uint16_t* __restrict__ W1t, const float* __restrict__ b1,
                   const uint16_t* __restrict__ W2t, const float* __restrict__ b2,
                   float* __restrict__ h2row,
                   float* __restrict__ bnSum, float* __restrict__ bnSq,
                   int lane) {
  const int row  = lane & 15;       // A-matrix row handled by this lane
  const int half = lane >> 4;       // lane-half select (ISA 7.12.2 layouts)
  const int colL = lane & 15;       // B/C/D column handled by this lane

  // ------------------------- layer 1: K1 -> N1, ELU -> bf16 LDS
  for (int nt = 0; nt < N1 / 16; ++nt) {
    v8f acc;
    const float bias = b1[nt * 16 + colL];
#pragma unroll
    for (int v = 0; v < 8; ++v) acc[v] = bias;
    if (nt + 1 < N1 / 16)  // keep the weight stream warm (global_prefetch_b8)
      __builtin_prefetch(W1t + (size_t)((nt + 1) * 16 + colL) * K1, 0, 0);
#pragma unroll
    for (int ks = 0; ks < K1 / 32; ++ks) {
      FragBF a, b;
      const uint16_t* ab = xin + row * K1 + ks * 32 + half * 8;
#pragma unroll
      for (int v = 0; v < 4; ++v) {
        a.u[v]     = *(const uint32_t*)(ab + 2 * v);        // K = kb+kh+2v
        a.u[v + 4] = *(const uint32_t*)(ab + 16 + 2 * v);   // K = kb+kh+16+2v
      }
      const uint16_t* bb = W1t + (size_t)(nt * 16 + colL) * K1 + ks * 32 + half * 8;
#pragma unroll
      for (int v = 0; v < 4; ++v) {
        b.u[v]     = *(const uint32_t*)(bb + 2 * v);
        b.u[v + 4] = *(const uint32_t*)(bb + 16 + 2 * v);
      }
      acc = __builtin_amdgcn_wmma_f32_16x16x32_bf16(false, a.v, false, b.v,
                                                    (short)0, acc, false, false);
    }
#pragma unroll
    for (int v = 0; v < 8; ++v) {
      const int m = v + half * 8;                 // D layout: vgpr v -> row m
      h1[m * N1 + nt * 16 + colL] = f2bf(elu(acc[v]));
    }
  }

  // ------------------------- layer 2: N1 -> N2, ELU -> f32 global + BN stats
  for (int nt = 0; nt < N2 / 16; ++nt) {
    v8f acc;
    const float bias = b2[nt * 16 + colL];
#pragma unroll
    for (int v = 0; v < 8; ++v) acc[v] = bias;
#pragma unroll
    for (int ks = 0; ks < N1 / 32; ++ks) {
      FragBF a, b;
      const uint16_t* ab = h1 + row * N1 + ks * 32 + half * 8;
#pragma unroll
      for (int v = 0; v < 4; ++v) {
        a.u[v]     = *(const uint32_t*)(ab + 2 * v);
        a.u[v + 4] = *(const uint32_t*)(ab + 16 + 2 * v);
      }
      const uint16_t* bb = W2t + (size_t)(nt * 16 + colL) * N1 + ks * 32 + half * 8;
#pragma unroll
      for (int v = 0; v < 4; ++v) {
        b.u[v]     = *(const uint32_t*)(bb + 2 * v);
        b.u[v + 4] = *(const uint32_t*)(bb + 16 + 2 * v);
      }
      acc = __builtin_amdgcn_wmma_f32_16x16x32_bf16(false, a.v, false, b.v,
                                                    (short)0, acc, false, false);
    }
    float s = 0.f, q = 0.f;
#pragma unroll
    for (int v = 0; v < 8; ++v) {
      const int m = v + half * 8;
      const float val = elu(acc[v]);
      h2row[(size_t)m * N2 + nt * 16 + colL] = val;
      s += val;
      q += val * val;
    }
    atomicAdd(&bnSum[nt * 16 + colL], s);
    atomicAdd(&bnSq [nt * 16 + colL], q);
  }
}

// ---------------------------------------------------------------------------
// Weight prep: fp32 [IN][OUT] -> bf16 [OUT][IN]  (K contiguous for B-frags)
// ---------------------------------------------------------------------------
__global__ void prep_w_kernel(const float* __restrict__ src,
                              uint16_t* __restrict__ dst, int IN, int OUT) {
  const int idx = blockIdx.x * blockDim.x + threadIdx.x;
  if (idx >= IN * OUT) return;
  const int i = idx / OUT, o = idx - i * OUT;
  dst[(size_t)o * IN + i] = f2bf(src[idx]);
}

__global__ void zero_stats_kernel(float* __restrict__ stats) {
  if (threadIdx.x < 256) stats[threadIdx.x] = 0.f;
}

// mean/var over cnt rows -> scale/shift at stats[256..383]/[384..511]
__global__ void bn_finalize_kernel(float* __restrict__ stats, float cnt,
                                   const float* __restrict__ gamma,
                                   const float* __restrict__ beta) {
  const int t = threadIdx.x;
  if (t >= 128) return;
  const float mean = stats[t] / cnt;
  const float var  = stats[128 + t] / cnt - mean * mean;
  const float sc   = gamma[t] * __frsqrt_rn(var + BN_EPS);
  stats[256 + t] = sc;
  stats[384 + t] = beta[t] - mean * sc;
}

// ---------------------------------------------------------------------------
// Edge MLP: gather (send||recv) per edge row, fused 256->256->128 WMMA MLP.
// ---------------------------------------------------------------------------
__global__ __launch_bounds__(WAVES * 32)
void edge_mlp_kernel(const float* __restrict__ node_src,
                     const uint16_t* __restrict__ W1t, const float* __restrict__ b1,
                     const uint16_t* __restrict__ W2t, const float* __restrict__ b2,
                     float* __restrict__ h2, float* __restrict__ gSum,
                     float* __restrict__ gSq) {
  __shared__ __align__(16) uint16_t xin[WAVES * 16 * 256];
  __shared__ __align__(16) uint16_t h1 [WAVES * 16 * 256];
  __shared__ float bnSum[EOc], bnSq[EOc];

  const int tid = threadIdx.x, wave = tid >> 5, lane = tid & 31;
  if (tid < EOc) { bnSum[tid] = 0.f; bnSq[tid] = 0.f; }
  __syncthreads();

  const int r0 = (blockIdx.x * WAVES + wave) * 16;   // first of 16 edge rows
  uint16_t* xw = xin + wave * 16 * 256;

  // gather: x[row] = concat(node[send], node[recv]) converted to bf16
  for (int idx = lane; idx < 16 * 256; idx += 32) {
    const int rr = idx >> 8, k = idx & 255;
    const int r  = r0 + rr;
    const int b  = r / Ec, e = r - b * Ec;
    const int rcv = e / (Nc - 1);
    const int j   = e - rcv * (Nc - 1);
    const int snd = j + (j >= rcv);
    const int node = (k < Dc) ? snd : rcv;           // sends first, then recs
    xw[rr * 256 + k] = f2bf(node_src[((size_t)b * Nc + node) * Dc + (k & (Dc - 1))]);
  }
  // per-wave private LDS segment: same-wave DS ordering, no barrier needed
  mlp2_core<256, 256, 128>(xw, h1 + wave * 16 * 256, W1t, b1, W2t, b2,
                           h2 + (size_t)r0 * EOc, bnSum, bnSq, lane);
  __syncthreads();
  if (tid < EOc) { atomicAdd(&gSum[tid], bnSum[tid]); atomicAdd(&gSq[tid], bnSq[tid]); }
}

// ef = h2*scale + shift (+ skip accumulated in-place across passes)
__global__ void edge_apply_kernel(const float* __restrict__ h2,
                                  const float* __restrict__ scale,
                                  const float* __restrict__ shift,
                                  float* __restrict__ ef, int pass) {
  const size_t idx = (size_t)blockIdx.x * blockDim.x + threadIdx.x;
  if (idx >= (size_t)BE * EOc) return;
  const int c = (int)(idx & (EOc - 1));
  float v = h2[idx] * scale[c] + shift[c];
  if (pass) v += ef[idx];
  ef[idx] = v;
}

// incoming/wnorm: receiver n owns contiguous edges [63n, 63n+63) -> no atomics
__global__ __launch_bounds__(128)
void aggregate_kernel(const float* __restrict__ ef, const float* __restrict__ ew,
                      float* __restrict__ agg) {
  const int b = blockIdx.x / Nc, n = blockIdx.x - b * Nc, d = threadIdx.x;
  const size_t ebase = (size_t)b * Ec + (size_t)n * (Nc - 1);
  float wsum = 0.f, acc = 0.f;
#pragma unroll 7
  for (int t = 0; t < Nc - 1; ++t) {
    const float w = ew[ebase + t];
    wsum += w;
    acc  += ef[(ebase + t) * EOc + d] * w;
  }
  agg[((size_t)b * Nc + n) * NOc + d] = acc / wsum;
}

// ---------------------------------------------------------------------------
// Node MLP: fused 128->256->128 WMMA MLP on aggregated features.
// ---------------------------------------------------------------------------
__global__ __launch_bounds__(WAVES * 32)
void node_mlp_kernel(const float* __restrict__ agg,
                     const uint16_t* __restrict__ W1t, const float* __restrict__ b1,
                     const uint16_t* __restrict__ W2t, const float* __restrict__ b2,
                     float* __restrict__ h2, float* __restrict__ gSum,
                     float* __restrict__ gSq) {
  __shared__ __align__(16) uint16_t xin[WAVES * 16 * 128];
  __shared__ __align__(16) uint16_t h1 [WAVES * 16 * 256];
  __shared__ float bnSum[NOc], bnSq[NOc];

  const int tid = threadIdx.x, wave = tid >> 5, lane = tid & 31;
  if (tid < NOc) { bnSum[tid] = 0.f; bnSq[tid] = 0.f; }
  __syncthreads();

  const int r0 = (blockIdx.x * WAVES + wave) * 16;
  uint16_t* xw = xin + wave * 16 * 128;
  for (int idx = lane; idx < 16 * 128; idx += 32) {
    const int rr = idx >> 7, k = idx & 127;
    xw[rr * 128 + k] = f2bf(agg[(size_t)(r0 + rr) * 128 + k]);
  }
  mlp2_core<128, 256, 128>(xw, h1 + wave * 16 * 256, W1t, b1, W2t, b2,
                           h2 + (size_t)r0 * NOc, bnSum, bnSq, lane);
  __syncthreads();
  if (tid < NOc) { atomicAdd(&gSum[tid], bnSum[tid]); atomicAdd(&gSq[tid], bnSq[tid]); }
}

__global__ void node_apply_kernel(const float* __restrict__ nsrc,
                                  const float* __restrict__ h2,
                                  const float* __restrict__ scale,
                                  const float* __restrict__ shift,
                                  float* __restrict__ ndst) {
  const size_t idx = (size_t)blockIdx.x * blockDim.x + threadIdx.x;
  if (idx >= (size_t)BNr * NOc) return;
  const int c = (int)(idx & (NOc - 1));
  ndst[idx] = nsrc[idx] + h2[idx] * scale[c] + shift[c];
}

__global__ void decode_kernel(const float* __restrict__ ef,
                              const float* __restrict__ dec_w,
                              const float* __restrict__ dec_b,
                              float* __restrict__ out2) {
  const size_t r = (size_t)blockIdx.x * blockDim.x + threadIdx.x;
  if (r >= (size_t)BE) return;
  float s = dec_b[0];
#pragma unroll 8
  for (int c = 0; c < EOc; ++c) s += ef[r * EOc + c] * dec_w[c];
  out2[r] = s;
}

// ---------------------------------------------------------------------------
extern "C" void kernel_launch(void* const* d_in, const int* in_sizes, int n_in,
                              void* d_out, int out_size, void* d_ws, size_t ws_size,
                              hipStream_t stream) {
  (void)in_sizes; (void)n_in; (void)out_size; (void)ws_size;
  const float* node_feature = (const float*)d_in[0];   // [B,N,128]
  const float* edge_weight  = (const float*)d_in[1];   // [B,E,1]
  // d_in[2]/d_in[3]: one-hot encoders — indices derived arithmetically instead
  const float* ew1 = (const float*)d_in[4];  const float* eb1 = (const float*)d_in[5];
  const float* ew2 = (const float*)d_in[6];  const float* eb2 = (const float*)d_in[7];
  const float* eg  = (const float*)d_in[8];  const float* ebt = (const float*)d_in[9];
  const float* nw1 = (const float*)d_in[10]; const float* nb1 = (const float*)d_in[11];
  const float* nw2 = (const float*)d_in[12]; const float* nb2 = (const float*)d_in[13];
  const float* ng  = (const float*)d_in[14]; const float* nbt = (const float*)d_in[15];
  const float* dec_w = (const float*)d_in[16];
  const float* dec_b = (const float*)d_in[17];

  // workspace layout
  uint16_t* ew1t = (uint16_t*)d_ws;            // [256][256] bf16
  uint16_t* ew2t = ew1t + 256 * 256;           // [128][256]
  uint16_t* nw1t = ew2t + 128 * 256;           // [256][128]
  uint16_t* nw2t = nw1t + 256 * 128;           // [128][256]
  float* stats   = (float*)(nw2t + 128 * 256); // 512: sum|sq|scale|shift
  float* h2buf   = stats + 512;                // [BE][128] (reused for node h2)
  float* efbuf   = h2buf + (size_t)BE * EOc;   // [BE][128] edge skip/feature
  float* aggbuf  = efbuf + (size_t)BE * EOc;   // [B*N][128]
  float* nodebuf = aggbuf + (size_t)BNr * NOc; // [B*N][128] intermediate nodes

  float* out_node = (float*)d_out;                       // [B,N,128]
  float* out_edge = out_node + (size_t)BNr * NOc;        // [B,E,1]

  // one-time (per call) weight transposition to bf16
  prep_w_kernel<<<(256 * 256 + 255) / 256, 256, 0, stream>>>(ew1, ew1t, 256, 256);
  prep_w_kernel<<<(256 * 128 + 255) / 256, 256, 0, stream>>>(ew2, ew2t, 256, 128);
  prep_w_kernel<<<(128 * 256 + 255) / 256, 256, 0, stream>>>(nw1, nw1t, 128, 256);
  prep_w_kernel<<<(256 * 128 + 255) / 256, 256, 0, stream>>>(nw2, nw2t, 256, 128);

  const int edgeBlocks = (int)(BE / (WAVES * 16));        // 4032
  const int nodeBlocks = BNr / (WAVES * 16);              // 64

  for (int pass = 0; pass < 2; ++pass) {
    const float* nsrc = pass ? nodebuf : node_feature;
    float*       ndst = pass ? out_node : nodebuf;

    zero_stats_kernel<<<1, 256, 0, stream>>>(stats);
    edge_mlp_kernel<<<edgeBlocks, WAVES * 32, 0, stream>>>(
        nsrc, ew1t, eb1, ew2t, eb2, h2buf, stats, stats + 128);
    bn_finalize_kernel<<<1, 128, 0, stream>>>(stats, (float)BE, eg, ebt);
    edge_apply_kernel<<<(int)((BE * EOc + 255) / 256), 256, 0, stream>>>(
        h2buf, stats + 256, stats + 384, efbuf, pass);

    aggregate_kernel<<<BNr, 128, 0, stream>>>(efbuf, edge_weight, aggbuf);

    zero_stats_kernel<<<1, 256, 0, stream>>>(stats);
    node_mlp_kernel<<<nodeBlocks, WAVES * 32, 0, stream>>>(
        aggbuf, nw1t, nb1, nw2t, nb2, h2buf, stats, stats + 128);
    bn_finalize_kernel<<<1, 128, 0, stream>>>(stats, (float)(BNr), ng, nbt);
    node_apply_kernel<<<(BNr * NOc + 255) / 256, 256, 0, stream>>>(
        nsrc, h2buf, stats + 256, stats + 384, ndst);
  }

  decode_kernel<<<(int)((BE + 255) / 256), 256, 0, stream>>>(
      efbuf, dec_w, dec_b, out_edge);
}
